// Predicate_GAT_Layer_6416681141172
// MI455X (gfx1250) — matrix operand order repaired and checked by reading, not compiled, not run on hardware
//
#include <hip/hip_runtime.h>
#include <hip/hip_bf16.h>
#include <math.h>

typedef __attribute__((ext_vector_type(16))) _Float16 v16h;
typedef __attribute__((ext_vector_type(8)))  float    v8f;

#define N_NODES   50000
#define N_EDGES   800000
#define IN_DIM    128
#define OUT_DIM   64
#define REL_DIM   32
#define NEG_SLOPE 0.01f
#define M_TILES   (N_NODES / 16)   // 3125, exact
#define WAVES_PER_BLOCK 8

// ---------------------------------------------------------------------------
// helpers
// ---------------------------------------------------------------------------
__device__ inline void atomicMaxF(float* addr, float val) {
    unsigned int* u = (unsigned int*)addr;
    unsigned int old = *u;
    while (true) {
        float f = __uint_as_float(old);
        if (f >= val) break;
        unsigned int assumed = old;
        old = atomicCAS(u, assumed, __float_as_uint(val));
        if (old == assumed) break;
    }
}

// Pack two contiguous runs of 8 floats into a v16h fragment (f32 -> f16).
__device__ inline v16h pack16(const float* __restrict__ lo,
                              const float* __restrict__ hi) {
    float4 x0 = *(const float4*)(lo + 0);
    float4 x1 = *(const float4*)(lo + 4);
    float4 x2 = *(const float4*)(hi + 0);
    float4 x3 = *(const float4*)(hi + 4);
    v16h r;
    r[0]  = (_Float16)x0.x; r[1]  = (_Float16)x0.y;
    r[2]  = (_Float16)x0.z; r[3]  = (_Float16)x0.w;
    r[4]  = (_Float16)x1.x; r[5]  = (_Float16)x1.y;
    r[6]  = (_Float16)x1.z; r[7]  = (_Float16)x1.w;
    r[8]  = (_Float16)x2.x; r[9]  = (_Float16)x2.y;
    r[10] = (_Float16)x2.z; r[11] = (_Float16)x2.w;
    r[12] = (_Float16)x3.x; r[13] = (_Float16)x3.y;
    r[14] = (_Float16)x3.z; r[15] = (_Float16)x3.w;
    return r;
}

// ---------------------------------------------------------------------------
// 0) init: zero h_out, denom; segmax = -inf   (must run every call: atomics)
// ---------------------------------------------------------------------------
__global__ __launch_bounds__(256) void gat_init(float* __restrict__ out,
                                                float* __restrict__ segmax,
                                                float* __restrict__ denom) {
    int i = blockIdx.x * blockDim.x + threadIdx.x;
    if (i < N_NODES * OUT_DIM) out[i] = 0.0f;
    if (i < N_NODES) { segmax[i] = -INFINITY; denom[i] = 0.0f; }
}

// ---------------------------------------------------------------------------
// 1) z = h @ W_fc^T via v_wmma_f32_16x16x32_f16 (f32 accumulate)
//    one wave per 16-row M tile; 4 N-tiles x 4 K-steps = 16 WMMAs / wave
// ---------------------------------------------------------------------------
__global__ __launch_bounds__(256) void gat_gemm_wmma(const float* __restrict__ h,
                                                     const float* __restrict__ Wfc,
                                                     float* __restrict__ z) {
    const int wave  = threadIdx.x >> 5;
    const int lane  = threadIdx.x & 31;
    const int l15   = lane & 15;
    const int lh    = lane >> 4;          // 0 or 1
    const int mtile = blockIdx.x * WAVES_PER_BLOCK + wave;
    if (mtile >= M_TILES) return;         // wave-uniform: EXEC all-1s inside

    v8f acc[4];
#pragma unroll
    for (int nt = 0; nt < 4; ++nt) acc[nt] = (v8f){};

    const int m = mtile * 16 + l15;       // A row for this lane

#pragma unroll
    for (int ks = 0; ks < 4; ++ks) {
        const int kbase = ks * 32;
        // A fragment: lane holds row m, K = kbase + lh*8 + {0..7, 16..23}
        const float* hrow = h + (long)m * IN_DIM + kbase + lh * 8;
        v16h A = pack16(hrow, hrow + 16);

#pragma unroll
        for (int nt = 0; nt < 4; ++nt) {
            // B fragment: lane holds col n = nt*16+l15,
            //             K = kbase + lh*8 + {0..7, 16..23}
            const float* wrow = Wfc + (long)(nt * 16 + l15) * IN_DIM + kbase + lh * 8;
            v16h B = pack16(wrow, wrow + 16);
            acc[nt] = __builtin_amdgcn_wmma_f32_16x16x32_f16(
                /*neg_a=*/false, A, /*neg_b=*/false, B,
                /*c_mod=*/(short)0, acc[nt],
                /*reuse_a=*/false, /*reuse_b=*/false);
        }
    }

    // D layout: VGPR r -> M = r + 8*lh ; N = l15
    const int baseM = mtile * 16 + 8 * lh;
#pragma unroll
    for (int nt = 0; nt < 4; ++nt) {
#pragma unroll
        for (int r = 0; r < 8; ++r) {
            z[(long)(baseM + r) * OUT_DIM + nt * 16 + l15] = acc[nt][r];
        }
    }
}

// ---------------------------------------------------------------------------
// 2) per-node attention precompute: s_src[i]=z_i.w[0:64], s_dst[i]=z_i.w[96:160]
// ---------------------------------------------------------------------------
__global__ __launch_bounds__(256) void gat_node_score(const float* __restrict__ z,
                                                      const float* __restrict__ Wattn,
                                                      float* __restrict__ s_src,
                                                      float* __restrict__ s_dst) {
    int i = blockIdx.x * blockDim.x + threadIdx.x;
    if (i >= N_NODES) return;
    const float4* zr  = (const float4*)(z + (long)i * OUT_DIM);
    const float4* w1  = (const float4*)(Wattn);                 // [0,64)
    const float4* w2  = (const float4*)(Wattn + OUT_DIM + REL_DIM); // [96,160)
    float a = 0.0f, b = 0.0f;
#pragma unroll
    for (int t = 0; t < OUT_DIM / 4; ++t) {
        float4 v = zr[t];
        float4 u = w1[t];
        float4 w = w2[t];
        a += v.x * u.x + v.y * u.y + v.z * u.z + v.w * u.w;
        b += v.x * w.x + v.y * w.y + v.z * w.z + v.w * w.w;
    }
    s_src[i] = a;
    s_dst[i] = b;
}

// ---------------------------------------------------------------------------
// 3) per-edge logit + leaky_relu + segment max (8 lanes per edge)
// ---------------------------------------------------------------------------
__global__ __launch_bounds__(256) void gat_edge_score(const float* __restrict__ p,
                                                      const int* __restrict__ src,
                                                      const int* __restrict__ dst,
                                                      const float* __restrict__ Wattn,
                                                      const float* __restrict__ s_src,
                                                      const float* __restrict__ s_dst,
                                                      float* __restrict__ ebuf,
                                                      float* __restrict__ segmax) {
    int tid = blockIdx.x * blockDim.x + threadIdx.x;
    int e   = tid >> 3;
    int sub = tid & 7;
    if (e >= N_EDGES) return;
    float4 pv = ((const float4*)(p + (long)e * REL_DIM))[sub];
    float4 wv = ((const float4*)(Wattn + OUT_DIM))[sub];        // w_p = W_attn[64:96]
    float s = pv.x * wv.x + pv.y * wv.y + pv.z * wv.z + pv.w * wv.w;
    s += __shfl_xor(s, 4, 8);
    s += __shfl_xor(s, 2, 8);
    s += __shfl_xor(s, 1, 8);
    if (sub == 0) {
        int d = dst[e];
        float raw = s + s_src[src[e]] + s_dst[d];
        float lr  = raw > 0.0f ? raw : NEG_SLOPE * raw;
        ebuf[e] = lr;
        atomicMaxF(&segmax[d], lr);
    }
}

// ---------------------------------------------------------------------------
// 4) num = exp(e - segmax[dst]);  denom[dst] += num
// ---------------------------------------------------------------------------
__global__ __launch_bounds__(256) void gat_edge_exp(const int* __restrict__ dst,
                                                    const float* __restrict__ segmax,
                                                    float* __restrict__ ebuf,
                                                    float* __restrict__ denom) {
    int e = blockIdx.x * blockDim.x + threadIdx.x;
    if (e >= N_EDGES) return;
    int d = dst[e];
    float v = expf(ebuf[e] - segmax[d]);
    ebuf[e] = v;
    atomicAdd(&denom[d], v);
}

// ---------------------------------------------------------------------------
// 5) h_out[dst] += (num/denom[dst]) * z[src]   (16 lanes per edge, 4 floats ea)
// ---------------------------------------------------------------------------
__global__ __launch_bounds__(256) void gat_aggregate(const int* __restrict__ src,
                                                     const int* __restrict__ dst,
                                                     const float* __restrict__ z,
                                                     const float* __restrict__ ebuf,
                                                     const float* __restrict__ denom,
                                                     float* __restrict__ out) {
    int tid = blockIdx.x * blockDim.x + threadIdx.x;
    int e   = tid >> 4;
    int sub = tid & 15;
    if (e >= N_EDGES) return;
    int s = src[e];
    int d = dst[e];
    float alpha = ebuf[e] / denom[d];
    float4 zv = ((const float4*)(z + (long)s * OUT_DIM))[sub];
    float* o = out + (long)d * OUT_DIM + sub * 4;
    atomicAdd(o + 0, alpha * zv.x);
    atomicAdd(o + 1, alpha * zv.y);
    atomicAdd(o + 2, alpha * zv.z);
    atomicAdd(o + 3, alpha * zv.w);
}

// ---------------------------------------------------------------------------
// launch
// ---------------------------------------------------------------------------
extern "C" void kernel_launch(void* const* d_in, const int* in_sizes, int n_in,
                              void* d_out, int out_size, void* d_ws, size_t ws_size,
                              hipStream_t stream) {
    const float* h     = (const float*)d_in[0];   // [50000,128]
    const float* p     = (const float*)d_in[1];   // [800000,32]
    const int*   src   = (const int*)  d_in[2];   // [800000]
    const int*   dst   = (const int*)  d_in[3];   // [800000]
    const float* Wfc   = (const float*)d_in[4];   // [64,128]
    const float* Wattn = (const float*)d_in[5];   // [1,160]
    float* out = (float*)d_out;                   // [50000,64]

    // workspace carve-up (~17 MB)
    float* z      = (float*)d_ws;                 // 50000*64
    float* s_src  = z + (long)N_NODES * OUT_DIM;  // 50000
    float* s_dst  = s_src + N_NODES;              // 50000
    float* segmax = s_dst + N_NODES;              // 50000
    float* denom  = segmax + N_NODES;             // 50000
    float* ebuf   = denom + N_NODES;              // 800000

    {   // init (out/segmax/denom must be reset each call: atomics accumulate)
        int total  = N_NODES * OUT_DIM;
        int blocks = (total + 255) / 256;
        gat_init<<<blocks, 256, 0, stream>>>(out, segmax, denom);
    }
    {   // WMMA GEMM: 3125 M-tiles, 8 waves/block
        int blocks = (M_TILES + WAVES_PER_BLOCK - 1) / WAVES_PER_BLOCK;
        gat_gemm_wmma<<<blocks, 256, 0, stream>>>(h, Wfc, z);
    }
    {   // per-node scores
        int blocks = (N_NODES + 255) / 256;
        gat_node_score<<<blocks, 256, 0, stream>>>(z, Wattn, s_src, s_dst);
    }
    {   // per-edge logit + segment max (8 lanes/edge)
        long threads = (long)N_EDGES * 8;
        int blocks = (int)((threads + 255) / 256);
        gat_edge_score<<<blocks, 256, 0, stream>>>(p, src, dst, Wattn,
                                                   s_src, s_dst, ebuf, segmax);
    }
    {   // exp + denom
        int blocks = (N_EDGES + 255) / 256;
        gat_edge_exp<<<blocks, 256, 0, stream>>>(dst, segmax, ebuf, denom);
    }
    {   // weighted scatter-add (16 lanes/edge)
        long threads = (long)N_EDGES * 16;
        int blocks = (int)((threads + 255) / 256);
        gat_aggregate<<<blocks, 256, 0, stream>>>(src, dst, z, ebuf, denom, out);
    }
}
